// Memory_attention_41918880809551
// MI455X (gfx1250) — matrix operand order repaired and checked by reading, not compiled, not run on hardware
//
#include <hip/hip_runtime.h>
#include <math.h>
#include <string.h>

// ---------------- types & helpers ----------------
typedef __bf16 bf16;
typedef bf16  v16bf __attribute__((ext_vector_type(16)));
typedef float v8f   __attribute__((ext_vector_type(8)));
typedef float        f32x4 __attribute__((ext_vector_type(4)));
typedef unsigned int u32x4 __attribute__((ext_vector_type(4)));
typedef unsigned int u32x2 __attribute__((ext_vector_type(2)));
typedef int          v4i   __attribute__((ext_vector_type(4)));

union FragU { v16bf v; u32x4 q[2]; };

__device__ __forceinline__ unsigned short f2bf_bits(float f) {
    unsigned u = __float_as_uint(f);
    unsigned r = (u + 0x7FFFu + ((u >> 16) & 1u)) >> 16;   // RNE
    return (unsigned short)r;
}
// pack two f32 -> packed bf16x2 dword; single v_cvt_pk_bf16_f32 when available
__device__ __forceinline__ unsigned pack_bf16x2(float lo, float hi) {
#if __has_builtin(__builtin_amdgcn_cvt_pk_bf16_f32)
    auto r = __builtin_amdgcn_cvt_pk_bf16_f32(lo, hi);
    unsigned u; __builtin_memcpy(&u, &r, 4); return u;
#else
    return (unsigned)f2bf_bits(lo) | ((unsigned)f2bf_bits(hi) << 16);
#endif
}

#if __has_builtin(__builtin_amdgcn_global_load_async_to_lds_b128)
#define HAVE_ASYNC_LDS 1
#else
#define HAVE_ASYNC_LDS 0
#endif

// problem constants
#define BATCH 2
#define N1   2048
#define CDIM 768
#define HEADS 12
#define HD   64
#define LPAD 2064        // 2049 valid keys, padded to multiple of 16
#define LVALID 2049
#define CC   192         // compress dim
#define KK   64          // top-k
#define SENT -3.0e38f
#define BIGI 0x7fffffff

// ---------------- 1) compress MLP + gate ----------------
__global__ __launch_bounds__(256) void compress_kernel(
    const float* __restrict__ memk_in, const float* __restrict__ memv_in,
    const float* __restrict__ Wc1, const float* __restrict__ bc1,
    const float* __restrict__ Wc2, const float* __restrict__ bc2,
    const float* __restrict__ Wg,  const float* __restrict__ bg,
    float* __restrict__ mk_out, float* __restrict__ mv_out)
{
    __shared__ float meanbuf[CDIM];
    __shared__ float h1[CC];
    __shared__ float rowk[CDIM];
    __shared__ float rowv[CDIM];
    __shared__ float red[256];
    __shared__ float gsh;
    const int b = blockIdx.x, tid = threadIdx.x;

    for (int pass = 0; pass < 2; ++pass) {
        const float* src = pass ? memv_in : memk_in;
        float* rowdst = pass ? rowv : rowk;
        for (int c = tid; c < CDIM; c += 256) {
            float s = 0.f;
            for (int m = 0; m < 64; ++m) s += src[((size_t)b * 64 + m) * CDIM + c];
            meanbuf[c] = s * (1.0f / 64.0f);
        }
        __syncthreads();
        for (int o = tid; o < CC; o += 256) {
            float acc = bc1[o];
            for (int i = 0; i < CDIM; ++i) acc = fmaf(meanbuf[i], Wc1[(size_t)i * CC + o], acc);
            h1[o] = 0.5f * acc * (1.0f + erff(acc * 0.70710678118654752f));  // exact GELU
        }
        __syncthreads();
        for (int o = tid; o < CDIM; o += 256) {
            float acc = bc2[o];
            for (int i = 0; i < CC; ++i) acc = fmaf(h1[i], Wc2[(size_t)i * CDIM + o], acc);
            rowdst[o] = acc;
        }
        __syncthreads();
    }
    float p = 0.f;
    for (int c = tid; c < CDIM; c += 256) p = fmaf(rowk[c], Wg[c], p);
    red[tid] = p;
    __syncthreads();
    for (int s = 128; s > 0; s >>= 1) {
        if (tid < s) red[tid] += red[tid + s];
        __syncthreads();
    }
    if (tid == 0) gsh = 1.0f / (1.0f + expf(-(red[0] + bg[0])));
    __syncthreads();
    const float g = gsh;
    for (int c = tid; c < CDIM; c += 256) {
        mk_out[(size_t)b * CDIM + c] = rowk[c] * g;
        mv_out[(size_t)b * CDIM + c] = rowv[c] * g;
    }
}

// ---------------- 2) generic WMMA GEMM: Y = X @ W + bias ----------------
// mode 0: Q -> bf16 (b,h,row,d)  N1 stride     mode 1: K -> bf16, LPAD stride
// mode 2: V -> f32, LPAD stride                mode 3: O -> f32 (b,row,col) -> d_out
__global__ __launch_bounds__(256) void wmma_gemm_kernel(
    const float* __restrict__ X, const float* __restrict__ W,
    const float* __restrict__ bias, void* __restrict__ dst,
    int rowsPerBatch, int mode)
{
    __shared__ unsigned short Xs[64][40];   // 64 rows x 32 K (bf16), stride 80B: 16B aligned, bank-stride 20
    __shared__ unsigned short Wt[64][40];   // TRANSPOSED: 64 cols x 32 K (bf16)
    const int tid = threadIdx.x, lane = tid & 31, wave = tid >> 5;
    const int row0 = blockIdx.x * 64, col0 = blockIdx.y * 64, bb = blockIdx.z;
    const float* Xb = X + (size_t)bb * rowsPerBatch * CDIM;

    const int sr = wave & 3;          // subtile row 0..3
    const int scB = (wave >> 2) * 2;  // two subtile cols
    v8f acc0 = {}, acc1 = {};

    for (int k0 = 0; k0 < CDIM; k0 += 32) {
        // stage X chunk 64x32 (f32x4 loads -> v_cvt_pk_bf16_f32 -> ds_store_b64)
#pragma unroll
        for (int it = 0; it < 2; ++it) {
            int t = tid + it * 256;
            int r = t >> 3, c4 = (t & 7) * 4;
            f32x4 x = *(const f32x4*)&Xb[(size_t)(row0 + r) * CDIM + k0 + c4];
            u32x2 p;
            p.x = pack_bf16x2(x.x, x.y);
            p.y = pack_bf16x2(x.z, x.w);
            *(u32x2*)&Xs[r][c4] = p;
        }
        // stage W chunk 32x64 transposed -> Wt[col][k]
#pragma unroll
        for (int it = 0; it < 2; ++it) {
            int t = tid + it * 256;
            int kr = t >> 4, c4 = (t & 15) * 4;
            f32x4 wv = *(const f32x4*)&W[(size_t)(k0 + kr) * CDIM + col0 + c4];
            Wt[c4 + 0][kr] = f2bf_bits(wv.x);
            Wt[c4 + 1][kr] = f2bf_bits(wv.y);
            Wt[c4 + 2][kr] = f2bf_bits(wv.z);
            Wt[c4 + 3][kr] = f2bf_bits(wv.w);
        }
        if (k0 + 32 < CDIM)  // global_prefetch_b8 hint for next X chunk
            __builtin_prefetch(&Xb[(size_t)(row0 + (tid & 63)) * CDIM + k0 + 32], 0, 0);
        __syncthreads();

        // A fragment: two contiguous 16B runs per lane (ISA 7.12.2 16-bit A layout)
        const int m = sr * 16 + (lane & 15), hi = lane >> 4;
        FragU a;
        a.q[0] = *(const u32x4*)&Xs[m][8 * hi];
        a.q[1] = *(const u32x4*)&Xs[m][16 + 8 * hi];
        // B fragments: two contiguous 16B runs per lane from transposed W (same k order as A)
        const int n = lane & 15, kh = lane >> 4;
        FragU b0, b1;
        b0.q[0] = *(const u32x4*)&Wt[scB * 16 + n][8 * kh];
        b0.q[1] = *(const u32x4*)&Wt[scB * 16 + n][16 + 8 * kh];
        b1.q[0] = *(const u32x4*)&Wt[(scB + 1) * 16 + n][8 * kh];
        b1.q[1] = *(const u32x4*)&Wt[(scB + 1) * 16 + n][16 + 8 * kh];

        acc0 = __builtin_amdgcn_wmma_f32_16x16x32_bf16(false, a.v, false, b0.v, (short)0, acc0, false, false);
        acc1 = __builtin_amdgcn_wmma_f32_16x16x32_bf16(false, a.v, false, b1.v, (short)0, acc1, false, false);
        __syncthreads();
    }

    // epilogue (C/D layout: VGPR r -> M = r (lanes<16) / r+8; N = lane&15)
    const int n = lane & 15;
    for (int half = 0; half < 2; ++half) {
        v8f acc = half ? acc1 : acc0;
        int colg = col0 + (scB + half) * 16 + n;
        float bv = bias[colg];
#pragma unroll
        for (int r = 0; r < 8; ++r) {
            int rowg = row0 + sr * 16 + ((lane < 16) ? r : r + 8);
            float val = acc[r] + bv;
            if (mode == 0) {
                int h = colg >> 6, d = colg & 63;
                ((unsigned short*)dst)[(((size_t)bb * HEADS + h) * N1 + rowg) * HD + d] = f2bf_bits(val);
            } else if (mode == 1) {
                int h = colg >> 6, d = colg & 63;
                ((unsigned short*)dst)[(((size_t)bb * HEADS + h) * LPAD + rowg) * HD + d] = f2bf_bits(val);
            } else if (mode == 2) {
                int h = colg >> 6, d = colg & 63;
                ((float*)dst)[(((size_t)bb * HEADS + h) * LPAD + rowg) * HD + d] = val;
            } else {
                ((float*)dst)[((size_t)bb * N1 + rowg) * CDIM + colg] = val;
            }
        }
    }
}

// ---------------- 3) append gated memory row + zero padding ----------------
__global__ void append_mem_kernel(const float* __restrict__ mk, const float* __restrict__ mv,
                                  unsigned short* __restrict__ kb, float* __restrict__ vf)
{
    int blk = blockIdx.x;
    int b = blk / HEADS, h = blk % HEADS, d = threadIdx.x;   // 64 threads
    size_t bh = (size_t)b * HEADS + h;
    kb[(bh * LPAD + 2048) * HD + d] = f2bf_bits(mk[(size_t)b * CDIM + h * HD + d]);
    vf[(bh * LPAD + 2048) * HD + d] = mv[(size_t)b * CDIM + h * HD + d];
    for (int r = LVALID; r < LPAD; ++r) {
        kb[(bh * LPAD + r) * HD + d] = 0;
        vf[(bh * LPAD + r) * HD + d] = 0.f;
    }
}

// ---------------- 4) scores (WMMA) + top-64 + softmax + sparse AV ----------------
__global__ __launch_bounds__(256) void attn_topk_kernel(
    const unsigned short* __restrict__ qb, const unsigned short* __restrict__ kb,
    const float* __restrict__ vf, float* __restrict__ mid)
{
    __shared__ float S[16][2080];            // ~133KB score stripe in 320KB WGP LDS
    __shared__ float TV[16][KK];
    __shared__ int   TI[16][KK];
    __shared__ unsigned short QS[16][72];    // staged q tile, stride 144B (16B aligned, bank-stride 36)

    const int tid = threadIdx.x, lane = tid & 31, wave = tid >> 5;
    const int qt = blockIdx.x, h = blockIdx.y, b = blockIdx.z;
    const size_t bh = (size_t)b * HEADS + h;

    // stage Q tile (16 rows x 64 bf16 = 2KB) into LDS; async path on gfx1250
    if (tid < 128) {
        int r = tid >> 3, c8 = (tid & 7) * 8;
        const unsigned short* src = qb + (bh * N1 + (size_t)qt * 16 + r) * HD + c8;
#if HAVE_ASYNC_LDS
        __builtin_amdgcn_global_load_async_to_lds_b128(
            (__attribute__((address_space(1))) v4i*)src,
            (__attribute__((address_space(3))) v4i*)&QS[r][c8], 0, 0);
#else
        *(u32x4*)&QS[r][c8] = *(const u32x4*)src;
#endif
    }
#if HAVE_ASYNC_LDS
    asm volatile("s_wait_asynccnt 0x0" ::: "memory");
#endif
    __syncthreads();

    // A fragments (dims 0..31 and 32..63) via contiguous b128 LDS reads
    const int m = lane & 15, hi = lane >> 4;
    FragU a0, a1;
    a0.q[0] = *(const u32x4*)&QS[m][8 * hi];
    a0.q[1] = *(const u32x4*)&QS[m][16 + 8 * hi];
    a1.q[0] = *(const u32x4*)&QS[m][32 + 8 * hi];
    a1.q[1] = *(const u32x4*)&QS[m][48 + 8 * hi];

    // scores: 129 key tiles over 8 waves; B fragments are two 16B global reads per lane
    const int n = lane & 15, kh = lane >> 4;
    for (int kt = wave; kt < LPAD / 16; kt += 8) {
        const unsigned short* krow = kb + (bh * LPAD + (size_t)kt * 16 + n) * HD;
        FragU b0, b1;
        b0.q[0] = *(const u32x4*)(krow + 8 * kh);
        b0.q[1] = *(const u32x4*)(krow + 16 + 8 * kh);
        b1.q[0] = *(const u32x4*)(krow + 32 + 8 * kh);
        b1.q[1] = *(const u32x4*)(krow + 48 + 8 * kh);
        v8f c = {};
        c = __builtin_amdgcn_wmma_f32_16x16x32_bf16(false, a0.v, false, b0.v, (short)0, c, false, false);
        c = __builtin_amdgcn_wmma_f32_16x16x32_bf16(false, a1.v, false, b1.v, (short)0, c, false, false);
        int keyg = kt * 16 + n;
#pragma unroll
        for (int r = 0; r < 8; ++r) {
            int mm = (lane < 16) ? r : r + 8;
            float s = c[r] * 0.125f;            // scale = hd^-0.5
            if (keyg >= LVALID) s = -1e30f;     // mask pad keys
            S[mm][keyg] = s;
        }
    }
    __syncthreads();

    // each wave owns 2 query rows
    for (int rr = 0; rr < 2; ++rr) {
        const int mr = wave * 2 + rr;

        // ---- top-64 with per-lane exact top-2 cache of its strided subset ----
        float m1v = SENT, m2v = SENT; int m1i = BIGI, m2i = BIGI;
        for (int col = lane; col < LPAD; col += 32) {
            float v = S[mr][col];
            if (v > m1v) { m2v = m1v; m2i = m1i; m1v = v; m1i = col; }
            else if (v > m2v) { m2v = v; m2i = col; }
        }
        for (int t = 0; t < KK; ++t) {
            float best = m1v; int bi = m1i;
#pragma unroll
            for (int off = 16; off > 0; off >>= 1) {
                float ov = __shfl_xor(best, off, 32);
                int   oi = __shfl_xor(bi, off, 32);
                if (ov > best || (ov == best && oi < bi)) { best = ov; bi = oi; }
            }
            // winner lane (columns are lane-strided) consumes + updates its cache
            if (bi != BIGI && (bi & 31) == lane) {
                S[mr][bi] = SENT;
                m1v = m2v; m1i = m2i;
                m2v = SENT; m2i = BIGI;
                if (m1i == BIGI) {       // cache exhausted -> exact top-2 rescan
                    for (int col = lane; col < LPAD; col += 32) {
                        float v = S[mr][col];
                        if (v > m1v) { m2v = m1v; m2i = m1i; m1v = v; m1i = col; }
                        else if (v > m2v) { m2v = v; m2i = col; }
                    }
                }
            }
            if (lane == 0) { TV[mr][t] = best; TI[mr][t] = bi; }
        }

        // ---- softmax over the 64 selected ----
        float maxv = TV[mr][0];
        float e0 = __expf(TV[mr][lane] - maxv);
        float e1 = __expf(TV[mr][lane + 32] - maxv);
        float s = e0 + e1;
#pragma unroll
        for (int off = 16; off > 0; off >>= 1) s += __shfl_xor(s, off, 32);
        float inv = 1.0f / s;
        TV[mr][lane] = e0 * inv;
        TV[mr][lane + 32] = e1 * inv;

        // ---- sparse attn @ V: gather 64 value rows ----
        float acc0 = 0.f, acc1 = 0.f;
        for (int j = 0; j < KK; ++j) {
            float p = TV[mr][j];
            int idx = TI[mr][j];
            const float* vr = vf + (bh * LPAD + idx) * HD;
            acc0 = fmaf(p, vr[lane], acc0);
            acc1 = fmaf(p, vr[lane + 32], acc1);
        }
        float* dst = mid + ((size_t)b * N1 + (size_t)qt * 16 + mr) * CDIM + h * HD;
        dst[lane] = acc0;
        dst[lane + 32] = acc1;
    }
}

// ---------------- launch ----------------
extern "C" void kernel_launch(void* const* d_in, const int* in_sizes, int n_in,
                              void* d_out, int out_size, void* d_ws, size_t ws_size,
                              hipStream_t stream) {
    const float* x1   = (const float*)d_in[0];
    const float* x2   = (const float*)d_in[1];
    const float* memk = (const float*)d_in[2];
    const float* memv = (const float*)d_in[3];
    const float* Wq = (const float*)d_in[4];  const float* bq = (const float*)d_in[5];
    const float* Wk = (const float*)d_in[6];  const float* bk = (const float*)d_in[7];
    const float* Wv = (const float*)d_in[8];  const float* bv = (const float*)d_in[9];
    const float* Wc1 = (const float*)d_in[10]; const float* bc1 = (const float*)d_in[11];
    const float* Wc2 = (const float*)d_in[12]; const float* bc2 = (const float*)d_in[13];
    const float* Wg = (const float*)d_in[14];  const float* bg = (const float*)d_in[15];
    const float* Wp = (const float*)d_in[16];  const float* bp = (const float*)d_in[17];
    // perfix == 1 (static path)

    char* ws = (char*)d_ws;
    const size_t QB_BYTES  = (size_t)BATCH * HEADS * N1   * HD * 2;
    const size_t KB_BYTES  = (size_t)BATCH * HEADS * LPAD * HD * 2;
    const size_t VF_BYTES  = (size_t)BATCH * HEADS * LPAD * HD * 4;
    const size_t MID_BYTES = (size_t)BATCH * N1 * CDIM * 4;
    unsigned short* qb = (unsigned short*)ws;
    unsigned short* kb = (unsigned short*)(ws + QB_BYTES);
    float* vf  = (float*)(ws + QB_BYTES + KB_BYTES);
    float* mid = (float*)(ws + QB_BYTES + KB_BYTES + VF_BYTES);
    float* mkb = (float*)(ws + QB_BYTES + KB_BYTES + VF_BYTES + MID_BYTES);
    float* mvb = mkb + BATCH * CDIM;

    compress_kernel<<<BATCH, 256, 0, stream>>>(memk, memv, Wc1, bc1, Wc2, bc2, Wg, bg, mkb, mvb);

    dim3 gg(N1 / 64, CDIM / 64, BATCH);
    wmma_gemm_kernel<<<gg, 256, 0, stream>>>(x1, Wq, bq, (void*)qb, N1, 0);
    wmma_gemm_kernel<<<gg, 256, 0, stream>>>(x2, Wk, bk, (void*)kb, N1, 1);
    wmma_gemm_kernel<<<gg, 256, 0, stream>>>(x2, Wv, bv, (void*)vf, N1, 2);

    append_mem_kernel<<<BATCH * HEADS, HD, 0, stream>>>(mkb, mvb, kb, vf);

    dim3 ga(N1 / 16, HEADS, BATCH);
    attn_topk_kernel<<<ga, 256, 0, stream>>>(qb, kb, vf, mid);

    wmma_gemm_kernel<<<gg, 256, 0, stream>>>(mid, Wp, bp, d_out, N1, 3);
}
